// MoE_LoRA_MLP_43130061586817
// MI455X (gfx1250) — matrix operand order repaired and checked by reading, not compiled, not run on hardware
//
#include <hip/hip_runtime.h>
#include <hip/hip_bf16.h>

typedef unsigned short u16;
typedef __attribute__((ext_vector_type(16))) __bf16 v16bf;
typedef __attribute__((ext_vector_type(8)))  float  v8f;

#define T_TOK 4096
#define D_IN  1024
#define H_HID 4096
#define NEXP  8
#define RANK  32
#define ER    (NEXP*RANK)       /* 256 */
#define SCALING (1.0f/32.0f)

// CDNA5 async global->LDS copy path (GLOBAL_LOAD_ASYNC_TO_LDS_B128 + ASYNCcnt).
// Set to 0 to fall back to the sync (global->VGPR->ds_store) path.
#define USE_ASYNC_COPY 1

__device__ __forceinline__ u16 f2bf(float f) {
  __hip_bfloat16 h = __float2bfloat16(f);
  return *reinterpret_cast<u16*>(&h);
}

// ---------------------------------------------------------------------------
// Router: logits -> softmax (routing), argmax (expert index), expert_choice.
// One wave per token (wave32); lanes stride K, shfl_xor reduction.
// ---------------------------------------------------------------------------
__global__ __launch_bounds__(256)
void router_kernel(const float* __restrict__ x, const float* __restrict__ rw,
                   const float* __restrict__ rb, float* __restrict__ routing,
                   float* __restrict__ echoice, int* __restrict__ eidx) {
  const int wave = threadIdx.x >> 5;
  const int lid  = threadIdx.x & 31;
  const int t    = blockIdx.x * 8 + wave;

  float p[NEXP];
#pragma unroll
  for (int e = 0; e < NEXP; ++e) p[e] = 0.0f;

  const float* xt = x + (size_t)t * D_IN;
  for (int k = lid; k < D_IN; k += 32) {
    const float xv = xt[k];
#pragma unroll
    for (int e = 0; e < NEXP; ++e) p[e] += xv * rw[e * D_IN + k];
  }
#pragma unroll
  for (int e = 0; e < NEXP; ++e) {
#pragma unroll
    for (int off = 16; off > 0; off >>= 1) p[e] += __shfl_xor(p[e], off, 32);
    p[e] += rb[e];
  }
  // argmax on logits, first-max wins (matches jnp.argmax)
  float mx = p[0]; int am = 0;
#pragma unroll
  for (int e = 1; e < NEXP; ++e) { if (p[e] > mx) { mx = p[e]; am = e; } }
  float ex[NEXP]; float s = 0.0f;
#pragma unroll
  for (int e = 0; e < NEXP; ++e) { ex[e] = expf(p[e] - mx); s += ex[e]; }
  const float inv = 1.0f / s;
  if (lid == 0) {
#pragma unroll
    for (int e = 0; e < NEXP; ++e) {
      const float r = ex[e] * inv;
      routing[(size_t)t * NEXP + e] = r;
      const float mk = (e == am) ? 1.0f : 0.0f;
      echoice[(size_t)t * NEXP + e] = (mk - r) + r;   // same float order as ref
    }
    eidx[t] = am;
  }
}

// ---------------------------------------------------------------------------
// fp32 -> bf16 conversion (contiguous)
// ---------------------------------------------------------------------------
__global__ void cvt_f32_bf16(const float* __restrict__ in, u16* __restrict__ out, int n) {
  const int i = blockIdx.x * blockDim.x + threadIdx.x;
  if (i < n) out[i] = f2bf(in[i]);
}

// B[e, n, r] -> Bcat[n, e*32 + r]  (bf16), so LoRA B-stage is one dense GEMM.
__global__ void permute_B_bf16(const float* __restrict__ in, u16* __restrict__ out, int N) {
  const int i = blockIdx.x * blockDim.x + threadIdx.x;
  if (i >= N * ER) return;
  const int n = i >> 8;
  const int er = i & 255;
  const int e = er >> 5;
  const int r = er & 31;
  out[i] = f2bf(in[((size_t)e * N + n) * RANK + r]);
}

// ---------------------------------------------------------------------------
// Fused bf16 WMMA GEMM:  C[M,N] = A0[M,K0] @ W0[N,K0]^T  (+ A1[M,K1] @ W1[N,K1]^T)
// Block: 128x128 tile, 256 threads = 8 wave32; wave computes 32x64 (2x4 WMMA).
// Double-buffered LDS, tiles filled with GLOBAL_LOAD_ASYNC_TO_LDS_B128 and
// published with s_wait_asynccnt + workgroup barrier.
// EPI 0: hmask  (add Ab per col, zero if col's expert != token expert, *1/R, bf16)
// EPI 1: + fc1_b + (1/R)*Bb_down[e_t], exact GELU, bf16 out
// EPI 2: + fc2_b + (1/R)*Bb_up[e_t], fp32 out
// ---------------------------------------------------------------------------
template<int EPI>
__global__ __launch_bounds__(256)
void gemm_wmma(const u16* __restrict__ A0, const u16* __restrict__ W0, int K0,
               const u16* __restrict__ A1, const u16* __restrict__ W1, int K1,
               const float* __restrict__ bias, const float* __restrict__ Bb,
               const int* __restrict__ eidx, void* __restrict__ outp, int N) {
  __shared__ alignas(16) u16 As[2][128][48];   // +16 u16 pad: 96B row stride
  __shared__ alignas(16) u16 Ws[2][128][48];

  const int tid   = threadIdx.x;
  const int wave  = tid >> 5;
  const int lid   = tid & 31;
  const int waveM = wave >> 1;              // 0..3 -> 32-row group
  const int waveN = wave & 1;               // 0..1 -> 64-col group
  const int rowBlk = blockIdx.y * 128;
  const int colBlk = blockIdx.x * 128;

  v8f acc[2][4];
  const v8f vzero = {0.f,0.f,0.f,0.f,0.f,0.f,0.f,0.f};
#pragma unroll
  for (int m = 0; m < 2; ++m)
#pragma unroll
    for (int n = 0; n < 4; ++n) acc[m][n] = vzero;

  const int trow  = tid >> 2;               // 0..63
  const int tseg  = (tid & 3) * 8;          // K segment (8 bf16 = 16B)
  const int klo   = (lid >> 4) * 8;         // A frag: lanes>=16 hold K 8..15,24..31
  const int khalf = (lid >> 4) * 16;        // unused; B frag also loads at klo (K 8..15,24..31)
  const int mrow  = lid & 15;

  const int steps0 = K0 >> 5;
  const int steps1 = K1 >> 5;
  const int total  = steps0 + steps1;

  // Issue the (async) fill of LDS buffer `buf` with K-step `s` of the fused
  // two-phase K range (phase 0: main GEMM, phase 1: LoRA B-stage).
  auto issue = [&](int s, int buf) {
    const u16* Ap; const u16* Wp; int K; int k0;
    if (s < steps0) { Ap = A0; Wp = W0; K = K0; k0 = s << 5; }
    else            { Ap = A1; Wp = W1; K = K1; k0 = (s - steps0) << 5; }
    const u16* gA = Ap + (size_t)(rowBlk + trow) * K + k0 + tseg;
    const u16* gW = Wp + (size_t)(colBlk + trow) * K + k0 + tseg;
#if USE_ASYNC_COPY
    // LDS operand = wave-relative LDS byte address (low 32 bits of generic ptr)
    const unsigned la0 = (unsigned)(size_t)&As[buf][trow][tseg];
    const unsigned la1 = (unsigned)(size_t)&As[buf][trow + 64][tseg];
    const unsigned lw0 = (unsigned)(size_t)&Ws[buf][trow][tseg];
    const unsigned lw1 = (unsigned)(size_t)&Ws[buf][trow + 64][tseg];
    asm volatile("global_load_async_to_lds_b128 %0, %1, off"
                 :: "v"(la0), "v"(gA) : "memory");
    asm volatile("global_load_async_to_lds_b128 %0, %1, off"
                 :: "v"(la1), "v"(gA + (size_t)64 * K) : "memory");
    asm volatile("global_load_async_to_lds_b128 %0, %1, off"
                 :: "v"(lw0), "v"(gW) : "memory");
    asm volatile("global_load_async_to_lds_b128 %0, %1, off"
                 :: "v"(lw1), "v"(gW + (size_t)64 * K) : "memory");
#else
    const uint4 a0 = *(const uint4*)gA;
    const uint4 a1 = *(const uint4*)(gA + (size_t)64 * K);
    const uint4 w0 = *(const uint4*)gW;
    const uint4 w1 = *(const uint4*)(gW + (size_t)64 * K);
    *(uint4*)&As[buf][trow][tseg]      = a0;
    *(uint4*)&As[buf][trow + 64][tseg] = a1;
    *(uint4*)&Ws[buf][trow][tseg]      = w0;
    *(uint4*)&Ws[buf][trow + 64][tseg] = w1;
#endif
  };

  issue(0, 0);
  int cur = 0;
  for (int s = 0; s < total; ++s) {
#if USE_ASYNC_COPY
    // Each wave waits for its own async stores, then the barrier publishes the
    // tile workgroup-wide (and guarantees prior reads of buf cur^1 are done).
    asm volatile("s_wait_asynccnt 0x0" ::: "memory");
#endif
    __syncthreads();
    if (s + 1 < total) issue(s + 1, cur ^ 1);   // overlap fill with WMMAs

    v16bf afr[2], bfr[4];
#pragma unroll
    for (int m = 0; m < 2; ++m) {
      union { uint4 q[2]; v16bf v; } u;
      const u16* p = &As[cur][waveM * 32 + m * 16 + mrow][0];
      u.q[0] = *(const uint4*)(p + klo);
      u.q[1] = *(const uint4*)(p + klo + 16);
      afr[m] = u.v;
    }
#pragma unroll
    for (int n = 0; n < 4; ++n) {
      union { uint4 q[2]; v16bf v; } u;
      const u16* p = &Ws[cur][waveN * 64 + n * 16 + mrow][0];
      u.q[0] = *(const uint4*)(p + klo);
      u.q[1] = *(const uint4*)(p + klo + 16);
      bfr[n] = u.v;
    }
#pragma unroll
    for (int m = 0; m < 2; ++m)
#pragma unroll
      for (int n = 0; n < 4; ++n)
        acc[m][n] = __builtin_amdgcn_wmma_f32_16x16x32_bf16(
            false, afr[m], false, bfr[n], (short)0, acc[m][n], false, false);
    cur ^= 1;
  }

  // Epilogue. C/D layout: VGPR i: lanes 0-15 -> M=i, N=lane; lanes 16-31 -> M=i+8.
  const int rl = (lid >> 4) * 8;
  const int cl = lid & 15;
#pragma unroll
  for (int m = 0; m < 2; ++m) {
#pragma unroll
    for (int n = 0; n < 4; ++n) {
      const int col = colBlk + waveN * 64 + n * 16 + cl;
#pragma unroll
      for (int i = 0; i < 8; ++i) {
        const int row = rowBlk + waveM * 32 + m * 16 + rl + i;
        float v = acc[m][n][i];
        if (EPI == 0) {
          v += bias[col];                      // Ab (flat [256])
          const int e = col >> 5;
          v = (eidx[row] == e) ? v * SCALING : 0.0f;
          ((u16*)outp)[(size_t)row * N + col] = f2bf(v);
        } else if (EPI == 1) {
          v += bias[col] + SCALING * Bb[(size_t)eidx[row] * N + col];
          const float g = 0.5f * v * (1.0f + erff(v * 0.70710678118654752f));
          ((u16*)outp)[(size_t)row * N + col] = f2bf(g);
        } else {
          v += bias[col] + SCALING * Bb[(size_t)eidx[row] * N + col];
          ((float*)outp)[(size_t)row * N + col] = v;
        }
      }
    }
  }
}

// ---------------------------------------------------------------------------
extern "C" void kernel_launch(void* const* d_in, const int* in_sizes, int n_in,
                              void* d_out, int out_size, void* d_ws, size_t ws_size,
                              hipStream_t stream) {
  (void)in_sizes; (void)n_in; (void)out_size; (void)ws_size;

  const float* x        = (const float*)d_in[0];
  const float* router_w = (const float*)d_in[1];
  const float* router_b = (const float*)d_in[2];
  const float* fc1_w    = (const float*)d_in[3];
  const float* fc1_b    = (const float*)d_in[4];
  const float* fc2_w    = (const float*)d_in[5];
  const float* fc2_b    = (const float*)d_in[6];
  const float* A_down   = (const float*)d_in[7];
  const float* Ab_down  = (const float*)d_in[8];
  const float* B_down   = (const float*)d_in[9];
  const float* Bb_down  = (const float*)d_in[10];
  const float* A_up     = (const float*)d_in[11];
  const float* Ab_up    = (const float*)d_in[12];
  const float* B_up     = (const float*)d_in[13];
  const float* Bb_up    = (const float*)d_in[14];

  float* out_up      = (float*)d_out;                       // [T, D]
  float* out_routing = out_up + (size_t)T_TOK * D_IN;       // [T, E]
  float* out_choice  = out_routing + (size_t)T_TOK * NEXP;  // [T, E]

  char* wp = (char*)d_ws;
  auto carve = [&](size_t bytes) {
    char* p = wp; wp += (bytes + 255) & ~(size_t)255; return (void*)p;
  };
  int* eidx   = (int*)carve((size_t)T_TOK * 4);
  u16* xb     = (u16*)carve((size_t)T_TOK * D_IN * 2);
  u16* fc1_wb = (u16*)carve((size_t)H_HID * D_IN * 2);
  u16* fc2_wb = (u16*)carve((size_t)D_IN * H_HID * 2);
  u16* AcatD  = (u16*)carve((size_t)ER * D_IN * 2);
  u16* BcatD  = (u16*)carve((size_t)H_HID * ER * 2);
  u16* AcatU  = (u16*)carve((size_t)ER * H_HID * 2);
  u16* BcatU  = (u16*)carve((size_t)D_IN * ER * 2);
  u16* hmaskD = (u16*)carve((size_t)T_TOK * ER * 2);
  u16* hmaskU = (u16*)carve((size_t)T_TOK * ER * 2);
  u16* a_bf   = (u16*)carve((size_t)T_TOK * H_HID * 2);

  router_kernel<<<T_TOK / 8, 256, 0, stream>>>(x, router_w, router_b,
                                               out_routing, out_choice, eidx);

  auto cvtN = [&](const float* in, u16* out, size_t n) {
    cvt_f32_bf16<<<dim3((unsigned)((n + 255) / 256)), 256, 0, stream>>>(in, out, (int)n);
  };
  cvtN(x,      xb,     (size_t)T_TOK * D_IN);
  cvtN(fc1_w,  fc1_wb, (size_t)H_HID * D_IN);
  cvtN(fc2_w,  fc2_wb, (size_t)D_IN * H_HID);
  cvtN(A_down, AcatD,  (size_t)ER * D_IN);    // [E,R,D] is already [256,D]
  cvtN(A_up,   AcatU,  (size_t)ER * H_HID);   // [E,R,H] is already [256,H]
  permute_B_bf16<<<(H_HID * ER + 255) / 256, 256, 0, stream>>>(B_down, BcatD, H_HID);
  permute_B_bf16<<<(D_IN * ER + 255) / 256, 256, 0, stream>>>(B_up, BcatU, D_IN);

  // hmask_down = mask .* (x @ Acat_down^T + Ab) * 1/R        [T, 256] bf16
  gemm_wmma<0><<<dim3(ER / 128, T_TOK / 128), 256, 0, stream>>>(
      xb, AcatD, D_IN, nullptr, nullptr, 0, Ab_down, nullptr, eidx, hmaskD, ER);
  // a = GELU( x@fc1^T + hmaskD@Bcat_down^T + fc1_b + (1/R)Bb_down[e_t] )  -> bf16
  gemm_wmma<1><<<dim3(H_HID / 128, T_TOK / 128), 256, 0, stream>>>(
      xb, fc1_wb, D_IN, hmaskD, BcatD, ER, fc1_b, Bb_down, eidx, a_bf, H_HID);
  // hmask_up = mask .* (a @ Acat_up^T + Ab) * 1/R            [T, 256] bf16
  gemm_wmma<0><<<dim3(ER / 128, T_TOK / 128), 256, 0, stream>>>(
      a_bf, AcatU, H_HID, nullptr, nullptr, 0, Ab_up, nullptr, eidx, hmaskU, ER);
  // up = a@fc2^T + hmaskU@Bcat_up^T + fc2_b + (1/R)Bb_up[e_t]  -> fp32 d_out
  gemm_wmma<2><<<dim3(D_IN / 128, T_TOK / 128), 256, 0, stream>>>(
      a_bf, fc2_wb, H_HID, hmaskU, BcatU, ER, fc2_b, Bb_up, eidx, out_up, D_IN);
}
